// ContextQueryAttention_54889682043286
// MI455X (gfx1250) — matrix operand order repaired and checked by reading, not compiled, not run on hardware
//
#include <hip/hip_runtime.h>

typedef __attribute__((ext_vector_type(16))) _Float16 v16h;
typedef __attribute__((ext_vector_type(8)))  _Float16 v8h;
typedef __attribute__((ext_vector_type(8)))  float    v8f;

// Problem constants (from reference: b=64, n=2048, m=256, d=768)
constexpr int B = 64, N = 2048, M = 256, D = 768;
constexpr int KT  = 32;        // WMMA K per iteration (16x16x32 f16)
constexpr int LDK = KT + 8;    // padded LDS row stride in halves (80B, 16B-aligned)
constexpr int NSLICE = 8;      // column-softmax partial slices over n

// ---- workspace layout (float entries), total ~187 MB ----
constexpr size_t OFF_CW    = 0;                                  // [B*N]  ctx @ w_c
constexpr size_t OFF_QW    = OFF_CW    + (size_t)B * N;          // [B*M]  qry @ w_q
constexpr size_t OFF_RMAX  = OFF_QW    + (size_t)B * M;          // [B*N]
constexpr size_t OFF_RSUM  = OFF_RMAX  + (size_t)B * N;          // [B*N]
constexpr size_t OFF_CMAXP = OFF_RSUM  + (size_t)B * N;          // [B*NSLICE*M]
constexpr size_t OFF_CSUMP = OFF_CMAXP + (size_t)B * NSLICE * M; // [B*NSLICE*M]
constexpr size_t OFF_CMAX  = OFF_CSUMP + (size_t)B * NSLICE * M; // [B*M]
constexpr size_t OFF_CISUM = OFF_CMAX  + (size_t)B * M;          // [B*M] (1/colsum)
constexpr size_t OFF_TMP   = OFF_CISUM + (size_t)B * M;          // [B*M*D]
constexpr size_t OFF_S     = OFF_TMP   + (size_t)B * M * D;      // [B*N*M]

// ---------------- fragment helpers (CDNA5 WMMA VGPR layouts) ----------------
__device__ inline v8f vzero() {
  v8f z;
#pragma unroll
  for (int i = 0; i < 8; ++i) z[i] = 0.0f;
  return z;
}

// A-matrix 16x32 f16: lane<16 -> M=lane, K{0..7,16..23}; lane>=16 -> M=lane-16, K{8..15,24..31}
__device__ inline v16h load_frag_A(const _Float16* base, int stride, int row0, int lane) {
  const int kb = (lane < 16) ? 0 : 8;
  const _Float16* p = base + (size_t)(row0 + (lane & 15)) * stride;
  v8h lo = *(const v8h*)(p + kb);
  v8h hi = *(const v8h*)(p + kb + 16);
  v16h a;
#pragma unroll
  for (int i = 0; i < 8; ++i) { a[i] = lo[i]; a[i + 8] = hi[i]; }
  return a;
}

// B-matrix 32x16 f16: lane<16 -> N=lane, K{0..7,16..23}; lane>=16 -> N=lane-16, K{8..15,24..31}
// LDS tile stored as [col][k] so each lane reads two runs of 8 contiguous halves.
__device__ inline v16h load_frag_B(const _Float16* base, int stride, int col0, int lane) {
  const int kb = (lane < 16) ? 0 : 8;
  const _Float16* p = base + (size_t)(col0 + (lane & 15)) * stride + kb;
  v8h lo = *(const v8h*)(p);
  v8h hi = *(const v8h*)(p + 16);
  v16h b;
#pragma unroll
  for (int i = 0; i < 8; ++i) { b[i] = lo[i]; b[i + 8] = hi[i]; }
  return b;
}

// ---------------- kernel 1: row biases cw = ctx@w_c, qw = qry@w_q ----------------
__global__ __launch_bounds__(256) void k_bias(const float* __restrict__ ctx,
                                              const float* __restrict__ qry,
                                              const float* __restrict__ W,
                                              float* __restrict__ cw,
                                              float* __restrict__ qw) {
  const int wid = threadIdx.x >> 5, lane = threadIdx.x & 31;
  const long id = (long)blockIdx.x * 8 + wid;
  const float* src; const float* w; float* dst;
  if (id < (long)B * N) { src = ctx + id * (long)D; w = W;     dst = cw + id; }
  else { long q = id - (long)B * N; src = qry + q * (long)D; w = W + D; dst = qw + q; }
  float s = 0.0f;
#pragma unroll
  for (int j = 0; j < D / 32; ++j) { int k = lane + 32 * j; s += src[k] * w[k]; }
#pragma unroll
  for (int off = 16; off; off >>= 1) s += __shfl_xor(s, off, 32);
  if (lane == 0) *dst = s;
}

// ---------------- kernel 2: S = bias + (ctx*w_cq) @ qry^T  (WMMA f16) ----------------
__global__ __launch_bounds__(256) void k_S(const float* __restrict__ ctx,
                                           const float* __restrict__ qry,
                                           const float* __restrict__ W,
                                           const float* __restrict__ cw,
                                           const float* __restrict__ qw,
                                           float* __restrict__ S) {
  __shared__ __align__(16) _Float16 As[128][LDK];  // [n-row][k]  (ctx * w_cq)
  __shared__ __align__(16) _Float16 Bs[128][LDK];  // [m-col][k]  (qry)
  const int tid = threadIdx.x;
  const int bb = blockIdx.x, n0 = blockIdx.y * 128, m0 = blockIdx.z * 128;
  const float* wcq = W + 2 * D;
  const int wid = tid >> 5, lane = tid & 31;
  const int wn = wid & 3, wm = wid >> 2;         // wave tile: 32n x 64m
  v8f acc[2][4];
#pragma unroll
  for (int it = 0; it < 2; ++it)
#pragma unroll
    for (int jt = 0; jt < 4; ++jt) acc[it][jt] = vzero();

  const int tr = tid >> 2;            // 0..63
  const int tk = (tid & 3) * 8;       // 0,8,16,24

  for (int kk = 0; kk < D; kk += KT) {
    __syncthreads();
#pragma unroll
    for (int rp = 0; rp < 128; rp += 64) {
      const int r = tr + rp;
      const float* sa = ctx + ((size_t)bb * N + n0 + r) * D + kk + tk;
      const float* sb = qry + ((size_t)bb * M + m0 + r) * D + kk + tk;
#pragma unroll
      for (int j = 0; j < 8; ++j) {
        As[r][tk + j] = (_Float16)(sa[j] * wcq[kk + tk + j]);
        Bs[r][tk + j] = (_Float16)sb[j];
      }
    }
    __syncthreads();
    v16h bf[4];
#pragma unroll
    for (int jt = 0; jt < 4; ++jt) bf[jt] = load_frag_B(&Bs[0][0], LDK, wm * 64 + jt * 16, lane);
#pragma unroll
    for (int it = 0; it < 2; ++it) {
      v16h a = load_frag_A(&As[0][0], LDK, wn * 32 + it * 16, lane);
#pragma unroll
      for (int jt = 0; jt < 4; ++jt)
        acc[it][jt] = __builtin_amdgcn_wmma_f32_16x16x32_f16(false, a, false, bf[jt],
                                                             (short)0, acc[it][jt], false, false);
    }
  }
  const int rowadd = (lane & 16) ? 8 : 0;
  const int ncol = lane & 15;
#pragma unroll
  for (int it = 0; it < 2; ++it)
#pragma unroll
    for (int jt = 0; jt < 4; ++jt) {
      const int mcol = m0 + wm * 64 + jt * 16 + ncol;
      const float qv = qw[(size_t)bb * M + mcol];
#pragma unroll
      for (int r = 0; r < 8; ++r) {
        const int nrow = n0 + wn * 32 + it * 16 + r + rowadd;
        S[((size_t)bb * N + nrow) * M + mcol] =
            acc[it][jt][r] + cw[(size_t)bb * N + nrow] + qv;
      }
    }
}

// ---------------- kernel 3: per-row softmax stats (axis=-1, m=256) ----------------
__global__ __launch_bounds__(256) void k_rowstats(const float* __restrict__ S,
                                                  float* __restrict__ rmax,
                                                  float* __restrict__ rsum) {
  const int wid = threadIdx.x >> 5, lane = threadIdx.x & 31;
  const size_t row = (size_t)blockIdx.x * 8 + wid;
  const float* p = S + row * M;
  float x[8];
#pragma unroll
  for (int j = 0; j < 8; ++j) x[j] = p[lane + 32 * j];
  float mx = x[0];
#pragma unroll
  for (int j = 1; j < 8; ++j) mx = fmaxf(mx, x[j]);
#pragma unroll
  for (int off = 16; off; off >>= 1) mx = fmaxf(mx, __shfl_xor(mx, off, 32));
  float s = 0.0f;
#pragma unroll
  for (int j = 0; j < 8; ++j) s += __expf(x[j] - mx);
#pragma unroll
  for (int off = 16; off; off >>= 1) s += __shfl_xor(s, off, 32);
  if (lane == 0) { rmax[row] = mx; rsum[row] = s; }
}

// ---------------- kernel 4: per-column (axis=1) online softmax partials ----------------
__global__ __launch_bounds__(256) void k_colstats(const float* __restrict__ S,
                                                  float* __restrict__ cmaxP,
                                                  float* __restrict__ csumP) {
  const int bb = blockIdx.x, p = blockIdx.y, m = threadIdx.x;
  float mx = -3.0e38f, s = 0.0f;
  const float* base = S + ((size_t)bb * N + (size_t)p * (N / NSLICE)) * M + m;
  for (int i = 0; i < N / NSLICE; ++i) {
    const float x = base[(size_t)i * M];
    if (x > mx) { s = s * __expf(mx - x) + 1.0f; mx = x; }
    else        { s += __expf(x - mx); }
  }
  const size_t o = ((size_t)bb * NSLICE + p) * M + m;
  cmaxP[o] = mx; csumP[o] = s;
}

__global__ __launch_bounds__(256) void k_colmerge(const float* __restrict__ cmaxP,
                                                  const float* __restrict__ csumP,
                                                  float* __restrict__ cmax,
                                                  float* __restrict__ cisum) {
  const int idx = blockIdx.x * 256 + threadIdx.x;   // 0 .. B*M-1
  const int bb = idx / M, m = idx % M;
  float mx = -3.0e38f;
#pragma unroll
  for (int p = 0; p < NSLICE; ++p)
    mx = fmaxf(mx, cmaxP[((size_t)bb * NSLICE + p) * M + m]);
  float s = 0.0f;
#pragma unroll
  for (int p = 0; p < NSLICE; ++p) {
    const size_t o = ((size_t)bb * NSLICE + p) * M + m;
    s += csumP[o] * __expf(cmaxP[o] - mx);
  }
  cmax[idx] = mx;
  cisum[idx] = 1.0f / s;
}

// ---------------- kernel 5: tmp = softmax_n(S)^T @ ctx  (k = n, WMMA) ----------------
__global__ __launch_bounds__(256) void k_tmp(const float* __restrict__ S,
                                             const float* __restrict__ ctx,
                                             const float* __restrict__ cmax,
                                             const float* __restrict__ cisum,
                                             float* __restrict__ tmp) {
  __shared__ __align__(16) _Float16 As2[128][LDK];  // [m][k=n]  exp(S^T - colmax)
  __shared__ __align__(16) _Float16 Bs2[128][LDK];  // [d][k=n]  ctx^T
  const int tid = threadIdx.x;
  const int bb = blockIdx.x, m0 = blockIdx.y * 128, d0 = blockIdx.z * 128;
  const int wid = tid >> 5, lane = tid & 31;
  const int wm = wid & 3, wd = wid >> 2;           // wave tile: 32m x 64d
  v8f acc[2][4];
#pragma unroll
  for (int it = 0; it < 2; ++it)
#pragma unroll
    for (int jt = 0; jt < 4; ++jt) acc[it][jt] = vzero();

  for (int n0 = 0; n0 < N; n0 += KT) {
    __syncthreads();
#pragma unroll
    for (int p = 0; p < 16; ++p) {
      const int e = p * 256 + tid;
      const int c = e & 127, kk = e >> 7;          // c = mm or dd, kk = 0..31
      const float x = S[((size_t)bb * N + n0 + kk) * M + m0 + c];
      As2[c][kk] = (_Float16)__expf(x - cmax[(size_t)bb * M + m0 + c]);
      Bs2[c][kk] = (_Float16)ctx[((size_t)bb * N + n0 + kk) * D + d0 + c];
    }
    __syncthreads();
    v16h bf[4];
#pragma unroll
    for (int jt = 0; jt < 4; ++jt) bf[jt] = load_frag_B(&Bs2[0][0], LDK, wd * 64 + jt * 16, lane);
#pragma unroll
    for (int it = 0; it < 2; ++it) {
      v16h a = load_frag_A(&As2[0][0], LDK, wm * 32 + it * 16, lane);
#pragma unroll
      for (int jt = 0; jt < 4; ++jt)
        acc[it][jt] = __builtin_amdgcn_wmma_f32_16x16x32_f16(false, a, false, bf[jt],
                                                             (short)0, acc[it][jt], false, false);
    }
  }
  const int rowadd = (lane & 16) ? 8 : 0;
  const int dcolb = lane & 15;
#pragma unroll
  for (int it = 0; it < 2; ++it)
#pragma unroll
    for (int jt = 0; jt < 4; ++jt) {
#pragma unroll
      for (int r = 0; r < 8; ++r) {
        const int mrow = m0 + wm * 32 + it * 16 + r + rowadd;
        const int dcol = d0 + wd * 64 + jt * 16 + dcolb;
        tmp[((size_t)bb * M + mrow) * D + dcol] =
            acc[it][jt][r] * cisum[(size_t)bb * M + mrow];
      }
    }
}

// ---------------- kernel 6: A = P1 @ qry, Bout = P1 @ tmp  (fused, k = m = 256) ----------------
__global__ __launch_bounds__(256) void k_AB(const float* __restrict__ S,
                                            const float* __restrict__ qry,
                                            const float* __restrict__ tmp,
                                            const float* __restrict__ rmax,
                                            const float* __restrict__ rsum,
                                            float* __restrict__ outA,
                                            float* __restrict__ outB) {
  __shared__ __align__(16) _Float16 As3[64][LDK];   // [n][k=m] exp(S - rowmax)
  __shared__ __align__(16) _Float16 Bq3[128][LDK];  // [d][k=m] qry^T
  __shared__ __align__(16) _Float16 Bt3[128][LDK];  // [d][k=m] tmp^T
  const int tid = threadIdx.x;
  const int bb = blockIdx.x, n0 = blockIdx.y * 64, d0 = blockIdx.z * 128;
  const int wid = tid >> 5, lane = tid & 31;
  const int wn = wid & 3, wd = wid >> 2;            // wave tile: 16n x 64d (x2 outputs)
  v8f accA[4], accB[4];
#pragma unroll
  for (int jt = 0; jt < 4; ++jt) { accA[jt] = vzero(); accB[jt] = vzero(); }

  for (int k0 = 0; k0 < M; k0 += KT) {
    __syncthreads();
#pragma unroll
    for (int p = 0; p < 8; ++p) {                   // A tile: 64x32, k-contiguous reads
      const int e = p * 256 + tid;
      const int kk = e & 31, nn = e >> 5;
      const float x = S[((size_t)bb * N + n0 + nn) * M + k0 + kk];
      As3[nn][kk] = (_Float16)__expf(x - rmax[(size_t)bb * N + n0 + nn]);
    }
#pragma unroll
    for (int p = 0; p < 16; ++p) {                  // B tiles: transpose loads, coalesced in d
      const int e = p * 256 + tid;
      const int dd = e & 127, kk = e >> 7;
      Bq3[dd][kk] = (_Float16)qry[((size_t)bb * M + k0 + kk) * D + d0 + dd];
      Bt3[dd][kk] = (_Float16)tmp[((size_t)bb * M + k0 + kk) * D + d0 + dd];
    }
    __syncthreads();
    v16h a = load_frag_A(&As3[0][0], LDK, wn * 16, lane);
#pragma unroll
    for (int jt = 0; jt < 4; ++jt) {
      v16h bq = load_frag_B(&Bq3[0][0], LDK, wd * 64 + jt * 16, lane);
      accA[jt] = __builtin_amdgcn_wmma_f32_16x16x32_f16(false, a, false, bq,
                                                        (short)0, accA[jt], false, false);
      v16h bt = load_frag_B(&Bt3[0][0], LDK, wd * 64 + jt * 16, lane);
      accB[jt] = __builtin_amdgcn_wmma_f32_16x16x32_f16(false, a, false, bt,
                                                        (short)0, accB[jt], false, false);
    }
  }
  const int rowadd = (lane & 16) ? 8 : 0;
  const int dcolb = lane & 15;
#pragma unroll
  for (int jt = 0; jt < 4; ++jt) {
#pragma unroll
    for (int r = 0; r < 8; ++r) {
      const int nrow = n0 + wn * 16 + r + rowadd;
      const int dcol = d0 + wd * 64 + jt * 16 + dcolb;
      const float invr = 1.0f / rsum[(size_t)bb * N + nrow];
      const size_t o = ((size_t)bb * N + nrow) * D + dcol;
      outA[o] = accA[jt][r] * invr;
      outB[o] = accB[jt][r] * invr;
    }
  }
}

// ---------------- host-side launcher ----------------
extern "C" void kernel_launch(void* const* d_in, const int* in_sizes, int n_in,
                              void* d_out, int out_size, void* d_ws, size_t ws_size,
                              hipStream_t stream) {
  const float* ctx = (const float*)d_in[0];   // [B,N,D]
  const float* qry = (const float*)d_in[1];   // [B,M,D]
  const float* W   = (const float*)d_in[2];   // [3D]
  float* out = (float*)d_out;                 // A then B, each [B,N,D]
  float* ws  = (float*)d_ws;

  float* cw    = ws + OFF_CW;
  float* qw    = ws + OFF_QW;
  float* rmax  = ws + OFF_RMAX;
  float* rsum  = ws + OFF_RSUM;
  float* cmaxP = ws + OFF_CMAXP;
  float* csumP = ws + OFF_CSUMP;
  float* cmax  = ws + OFF_CMAX;
  float* cisum = ws + OFF_CISUM;
  float* tmp   = ws + OFF_TMP;
  float* S     = ws + OFF_S;

  // 1. row biases (B*N + B*M rows, 8 waves per block)
  k_bias<<<(B * N + B * M) / 8, 256, 0, stream>>>(ctx, qry, W, cw, qw);
  // 2. S = bias + (ctx*w_cq) @ qry^T
  k_S<<<dim3(B, N / 128, M / 128), 256, 0, stream>>>(ctx, qry, W, cw, qw, S);
  // 3. softmax stats
  k_rowstats<<<(B * N) / 8, 256, 0, stream>>>(S, rmax, rsum);
  k_colstats<<<dim3(B, NSLICE), 256, 0, stream>>>(S, cmaxP, csumP);
  k_colmerge<<<(B * M) / 256, 256, 0, stream>>>(cmaxP, csumP, cmax, cisum);
  // 4. tmp = softmax_n(S)^T @ ctx
  k_tmp<<<dim3(B, M / 128, D / 128), 256, 0, stream>>>(S, ctx, cmax, cisum, tmp);
  // 5. fused A = P1 @ qry, B = P1 @ tmp
  k_AB<<<dim3(B, N / 64, D / 128), 256, 0, stream>>>(S, qry, tmp, rmax, rsum,
                                                     out, out + (size_t)B * N * D);
}
